// PerceptGNN_69002944578206
// MI455X (gfx1250) — matrix operand order repaired and checked by reading, not compiled, not run on hardware
//
#include <hip/hip_runtime.h>

typedef _Float16 v16h __attribute__((ext_vector_type(16)));
typedef _Float16 v8h  __attribute__((ext_vector_type(8)));
typedef float    v8f  __attribute__((ext_vector_type(8)));

static constexpr int BB = 8;     // batch
static constexpr int NN = 1024;  // nodes
static constexpr int HH = 8;     // heads

// ---- workspace layout (bytes); total ~25.5 MB ----
static constexpr size_t OFF_XA   = 0;                        // 8 MB  f32 activations ping
static constexpr size_t OFF_XB   = (size_t)8  << 20;         // 8 MB  f32 activations pong
static constexpr size_t OFF_X16  = (size_t)16 << 20;         // 4 MB  f16 layer input
static constexpr size_t OFF_W16  = (size_t)20 << 20;         // 128KB f16 packed weights
static constexpr size_t OFF_HP   = (size_t)21 << 20;         // 4 MB  f16 packed features (WMMA-B layout)
static constexpr size_t OFF_ASRC = OFF_HP + ((size_t)4 << 20);   // 256KB f32 [B,H,N]
static constexpr size_t OFF_ADST = OFF_ASRC + ((size_t)256 << 10);

__device__ __forceinline__ float lrelu02(float v) { return fmaxf(v, 0.2f * v); }

// ---------------------------------------------------------------------------
// f32 -> f16 convert
// ---------------------------------------------------------------------------
__global__ void k_cvt16(const float* __restrict__ in, _Float16* __restrict__ out, int n) {
    int i = blockIdx.x * 256 + threadIdx.x;
    if (i < n) out[i] = (_Float16)in[i];
}

// ---------------------------------------------------------------------------
// Pack W [Fin, HC] f32 -> WMMA B-operand layout, f16.
// Block (kt,nt) = 32x16, 512 halfs: half index = lane*16 + (k&15),
// lane = (n&15) | (k&16).  One coalesced v16h load per lane in the GEMM.
// ---------------------------------------------------------------------------
__global__ void k_packw(const float* __restrict__ W, _Float16* __restrict__ Wp,
                        int Fin, int HC) {
    int t = blockIdx.x * 256 + threadIdx.x;
    if (t >= Fin * HC) return;
    int k = t / HC, n = t % HC;
    int kt = k >> 5, nt = n >> 4;
    int lane = (n & 15) | (k & 16);
    Wp[(size_t)(kt * (HC >> 4) + nt) * 512 + lane * 16 + (k & 15)] = (_Float16)W[t];
}

// ---------------------------------------------------------------------------
// WMMA GEMM: h = X16 @ Wp.  One wave per 16x16 output tile, K-loop over Fin/32.
// Epilogue writes h directly in packed WMMA-B layout for the aggregation pass.
// ---------------------------------------------------------------------------
__global__ void __launch_bounds__(256) k_gemm(const _Float16* __restrict__ X16,
                                              const _Float16* __restrict__ Wp,
                                              _Float16* __restrict__ Hp,
                                              int Fin, int HC, int C) {
    const int wave = threadIdx.x >> 5, lane = threadIdx.x & 31;
    const int nt = blockIdx.y * 8 + wave;      // n-tile (16 cols)
    const int row0 = blockIdx.x * 16;          // m-tile (16 rows of B*N)
    const int arow = row0 + (lane & 15);
    const int kb0 = (lane & 16) ? 16 : 0;      // A-operand K sublane offset
    const _Float16* abase = X16 + (size_t)arow * Fin + kb0;
    const _Float16* bbase = Wp + (size_t)nt * 512 + (size_t)lane * 16;
    const int bstride = (HC >> 4) * 512;

    v8f acc = {0.f, 0.f, 0.f, 0.f, 0.f, 0.f, 0.f, 0.f};
    const int nkt = Fin >> 5;
    for (int kt = 0; kt < nkt; ++kt) {
        v8h a0 = *(const v8h*)(abase + kt * 32);        // K = kb0..kb0+7
        v8h a1 = *(const v8h*)(abase + kt * 32 + 8);    // K = kb0+8..kb0+15
        v16h A = __builtin_shufflevector(a0, a1, 0, 1, 2, 3, 4, 5, 6, 7,
                                                 8, 9, 10, 11, 12, 13, 14, 15);
        v16h Bm = *(const v16h*)(bbase + (size_t)kt * bstride);
        acc = __builtin_amdgcn_wmma_f32_16x16x32_f16(false, A, false, Bm,
                                                     (short)0, acc, false, false);
    }
    // epilogue -> packed feature layout [b, hd, jt, ct] blocks of 512 halfs
    const int b = row0 >> 10, n0 = row0 & 1023;
    const int jt = n0 >> 5, khigh = (n0 >> 4) & 1;
    const int col0 = nt * 16;
    const int hd = col0 / C;
    const int ct = (col0 % C) >> 4;
    const int lane_p = (lane & 15) | (khigh << 4);
    const size_t blk = ((((size_t)b * HH + hd) * 64 + jt) * (C >> 4) + ct) * 512;
    v8h hv;
#pragma unroll
    for (int r = 0; r < 8; ++r) hv[r] = (_Float16)acc[r];
    *(v8h*)(Hp + blk + lane_p * 16 + ((lane >> 4) << 3)) = hv;
}

// ---------------------------------------------------------------------------
// Per-node attention coefficients: asrc/adst [B,H,N]
// ---------------------------------------------------------------------------
__global__ void k_avec(const _Float16* __restrict__ Hp, const float* __restrict__ Asv,
                       const float* __restrict__ Adv, float* __restrict__ ASRC,
                       float* __restrict__ ADST, int C) {
    int t = blockIdx.x * 256 + threadIdx.x;  // t = (b*N + n)*H + hd, exactly B*N*H threads
    int hd = t & 7;
    int bn = t >> 3;
    int n = bn & 1023, b = bn >> 10;
    int jt = n >> 5, kh = n & 15, khigh = (n >> 4) & 1;
    size_t base = (((size_t)b * HH + hd) * 64 + jt) * (size_t)(C >> 4) * 512;
    float s = 0.f, d = 0.f;
    for (int c = 0; c < C; ++c) {
        int lane_p = (c & 15) | (khigh << 4);
        float h = (float)Hp[base + (size_t)(c >> 4) * 512 + lane_p * 16 + kh];
        s += h * Asv[hd * C + c];
        d += h * Adv[hd * C + c];
    }
    ASRC[((size_t)b * HH + hd) * NN + n] = s;
    ADST[((size_t)b * HH + hd) * NN + n] = d;
}

// ---------------------------------------------------------------------------
// Fused masked-softmax aggregation.  WG = (b, 16-row i-tile); wave = head.
// Single pass over adj: P tile built in registers in exact WMMA-A layout,
// denominator accumulated via WMMA against an all-ones B matrix.
// ---------------------------------------------------------------------------
template <int C, bool ELU>
__global__ void __launch_bounds__(256) k_aggr(const _Float16* __restrict__ Hp,
                                              const float* __restrict__ adj,
                                              const float* __restrict__ ASRC,
                                              const float* __restrict__ ADST,
                                              const float* __restrict__ bias,
                                              float* __restrict__ out) {
    const int b = blockIdx.x >> 6;
    const int i0 = (blockIdx.x & 63) << 4;
    const int hd = threadIdx.x >> 5, lane = threadIdx.x & 31;
    const float* asr = ASRC + ((size_t)b * HH + hd) * NN;
    const float* ads = ADST + ((size_t)b * HH + hd) * NN;

    // Amax over all j (tiny, L2-hot); monotone lrelu => m_i bounds every score.
    float am = -1e30f;
#pragma unroll 8
    for (int t = 0; t < 32; ++t) am = fmaxf(am, asr[lane + 32 * t]);
#pragma unroll
    for (int m = 16; m > 0; m >>= 1) am = fmaxf(am, __shfl_xor(am, m, 32));

    const float adv = ads[i0 + (lane & 15)];
    const float mi = lrelu02(am + adv);

    const float* arow = adj + ((size_t)b * NN + i0 + (lane & 15)) * NN;
    const int kb0 = (lane & 16) ? 16 : 0;
    const size_t hbase = (((size_t)b * HH + hd) * 64) * (size_t)(C >> 4) * 512;

    v8f acc0 = {0.f, 0.f, 0.f, 0.f, 0.f, 0.f, 0.f, 0.f};
    v8f acc1 = {0.f, 0.f, 0.f, 0.f, 0.f, 0.f, 0.f, 0.f};
    v8f zac  = {0.f, 0.f, 0.f, 0.f, 0.f, 0.f, 0.f, 0.f};
    v16h bones;
#pragma unroll
    for (int i = 0; i < 16; ++i) bones[i] = (_Float16)1.0f;

    for (int jt = 0; jt < 64; ++jt) {
        const int j0 = jt * 32;
        if (jt + 1 < 64) __builtin_prefetch(arow + j0 + 32 + kb0, 0, 1);
        float4 q0 = *(const float4*)(arow + j0 + kb0);
        float4 q1 = *(const float4*)(arow + j0 + kb0 + 4);
        float4 q2 = *(const float4*)(arow + j0 + kb0 + 8);
        float4 q3 = *(const float4*)(arow + j0 + kb0 + 12);
        float4 s0 = *(const float4*)(asr + j0 + kb0);
        float4 s1 = *(const float4*)(asr + j0 + kb0 + 4);
        float4 s2 = *(const float4*)(asr + j0 + kb0 + 8);
        float4 s3 = *(const float4*)(asr + j0 + kb0 + 12);

        float aq[16], sj[16];
        aq[0]=q0.x; aq[1]=q0.y; aq[2]=q0.z; aq[3]=q0.w;
        aq[4]=q1.x; aq[5]=q1.y; aq[6]=q1.z; aq[7]=q1.w;
        aq[8]=q2.x; aq[9]=q2.y; aq[10]=q2.z; aq[11]=q2.w;
        aq[12]=q3.x; aq[13]=q3.y; aq[14]=q3.z; aq[15]=q3.w;
        sj[0]=s0.x; sj[1]=s0.y; sj[2]=s0.z; sj[3]=s0.w;
        sj[4]=s1.x; sj[5]=s1.y; sj[6]=s1.z; sj[7]=s1.w;
        sj[8]=s2.x; sj[9]=s2.y; sj[10]=s2.z; sj[11]=s2.w;
        sj[12]=s3.x; sj[13]=s3.y; sj[14]=s3.z; sj[15]=s3.w;

        v16h A;
#pragma unroll
        for (int u = 0; u < 16; ++u) {
            float ls = lrelu02(sj[u] + adv);
            A[u] = (_Float16)(aq[u] * __expf(ls - mi));  // in [0,1]; mask==0 -> exactly 0
        }

        const size_t jb = hbase + (size_t)jt * (C >> 4) * 512 + (size_t)lane * 16;
        v16h B0 = *(const v16h*)(Hp + jb);
        acc0 = __builtin_amdgcn_wmma_f32_16x16x32_f16(false, A, false, B0,
                                                      (short)0, acc0, false, false);
        if (C == 32) {
            v16h B1 = *(const v16h*)(Hp + jb + 512);
            acc1 = __builtin_amdgcn_wmma_f32_16x16x32_f16(false, A, false, B1,
                                                          (short)0, acc1, false, false);
        }
        zac = __builtin_amdgcn_wmma_f32_16x16x32_f16(false, A, false, bones,
                                                     (short)0, zac, false, false);
    }

    // epilogue: out = acc/Z + bias, optional elu
    const int col = lane & 15;
    const float bia0 = bias[hd * C + col];
    const float bia1 = (C == 32) ? bias[hd * C + 16 + col] : 0.f;
#pragma unroll
    for (int r = 0; r < 8; ++r) {
        const int row = i0 + r + ((lane >> 4) << 3);
        const float inv = 1.0f / zac[r];
        float* orow = out + ((size_t)b * NN + row) * (HH * C) + hd * C;
        float v0 = acc0[r] * inv + bia0;
        if (ELU) v0 = v0 > 0.f ? v0 : (__expf(v0) - 1.f);
        orow[col] = v0;
        if (C == 32) {
            float v1 = acc1[r] * inv + bia1;
            if (ELU) v1 = v1 > 0.f ? v1 : (__expf(v1) - 1.f);
            orow[16 + col] = v1;
        }
    }
}

// ---------------------------------------------------------------------------
// Host-side layer driver
// ---------------------------------------------------------------------------
static void run_layer(const float* Xin, const float* adj, const float* W,
                      const float* asv, const float* adv, const float* bias,
                      float* Xout, int Fin, int C, bool elu, char* ws,
                      hipStream_t stream) {
    const int HC = HH * C;
    _Float16* X16 = (_Float16*)(ws + OFF_X16);
    _Float16* W16 = (_Float16*)(ws + OFF_W16);
    _Float16* Hp  = (_Float16*)(ws + OFF_HP);
    float* ASRC = (float*)(ws + OFF_ASRC);
    float* ADST = (float*)(ws + OFF_ADST);

    const int nx = BB * NN * Fin;
    k_cvt16<<<(nx + 255) / 256, 256, 0, stream>>>(Xin, X16, nx);
    k_packw<<<(Fin * HC + 255) / 256, 256, 0, stream>>>(W, W16, Fin, HC);
    k_gemm<<<dim3(BB * NN / 16, HC / 128), 256, 0, stream>>>(X16, W16, Hp, Fin, HC, C);
    k_avec<<<BB * NN * HH / 256, 256, 0, stream>>>(Hp, asv, adv, ASRC, ADST, C);
    dim3 ag(BB * (NN / 16));
    if (C == 32) {
        if (elu) k_aggr<32, true ><<<ag, 256, 0, stream>>>(Hp, adj, ASRC, ADST, bias, Xout);
        else     k_aggr<32, false><<<ag, 256, 0, stream>>>(Hp, adj, ASRC, ADST, bias, Xout);
    } else {
        if (elu) k_aggr<16, true ><<<ag, 256, 0, stream>>>(Hp, adj, ASRC, ADST, bias, Xout);
        else     k_aggr<16, false><<<ag, 256, 0, stream>>>(Hp, adj, ASRC, ADST, bias, Xout);
    }
}

extern "C" void kernel_launch(void* const* d_in, const int* in_sizes, int n_in,
                              void* d_out, int out_size, void* d_ws, size_t ws_size,
                              hipStream_t stream) {
    (void)in_sizes; (void)n_in; (void)out_size; (void)ws_size;
    const float* x   = (const float*)d_in[0];
    const float* adj = (const float*)d_in[1];
    const float* W1  = (const float*)d_in[2];
    const float* as1 = (const float*)d_in[3];
    const float* ad1 = (const float*)d_in[4];
    const float* b1  = (const float*)d_in[5];
    const float* W2  = (const float*)d_in[6];
    const float* as2 = (const float*)d_in[7];
    const float* ad2 = (const float*)d_in[8];
    const float* b2  = (const float*)d_in[9];
    const float* W3  = (const float*)d_in[10];
    const float* as3 = (const float*)d_in[11];
    const float* ad3 = (const float*)d_in[12];
    const float* b3  = (const float*)d_in[13];
    const float* W4  = (const float*)d_in[14];
    const float* as4 = (const float*)d_in[15];
    const float* ad4 = (const float*)d_in[16];
    const float* b4  = (const float*)d_in[17];

    char* ws = (char*)d_ws;
    float* XA = (float*)(ws + OFF_XA);
    float* XB = (float*)(ws + OFF_XB);

    run_layer(x,  adj, W1, as1, ad1, b1, XA,             64,  32, false, ws, stream);
    run_layer(XA, adj, W2, as2, ad2, b2, XB,             256, 32, true,  ws, stream);
    run_layer(XB, adj, W3, as3, ad3, b3, XA,             256, 16, false, ws, stream);
    run_layer(XA, adj, W4, as4, ad4, b4, (float*)d_out,  128, 16, true,  ws, stream);
}
